// SelfAttention_24223615550276
// MI455X (gfx1250) — matrix operand order repaired and checked by reading, not compiled, not run on hardware
//
#include <hip/hip_runtime.h>

// ---------------------------------------------------------------------------
// CDNA5 / gfx1250 self-attention (GroupNorm -> QKV -> flash attn -> proj+res)
// GEMMs: v_wmma_f32_16x16x32_bf16 (wave32), A-tiles staged in LDS via the
// Tensor Data Mover (tensor_load_to_lds + s_wait_tensorcnt), prefetch hints
// for streaming K/V tiles.
// ---------------------------------------------------------------------------

typedef __attribute__((ext_vector_type(16))) __bf16 v16bf;
typedef __attribute__((ext_vector_type(8)))  __bf16 v8bf;
typedef __attribute__((ext_vector_type(8)))  float  v8f;
typedef __attribute__((ext_vector_type(4)))  unsigned int v4u;
typedef __attribute__((ext_vector_type(8)))  int v8i;
typedef __attribute__((ext_vector_type(4)))  int v4i;

#define CDIM 512
#define NTOK 4096
#define NB   4
#define ATTN_SCALE 0.04419417382415922f  // 1/sqrt(512)

// ---- WMMA helpers ---------------------------------------------------------

__device__ inline v8f wmma_bf16(v16bf a, v16bf b, v8f c) {
  return __builtin_amdgcn_wmma_f32_16x16x32_bf16(
      /*neg_a=*/false, a, /*neg_b=*/false, b,
      /*c_mod=*/(short)0, c, /*reuse_a=*/false, /*reuse_b=*/false);
}

// A operand: 16 (M) x 32 (K) bf16 tile from row-major [rows x ld] source.
__device__ inline v16bf load_frag_a(const __bf16* base, int ld, int k0, int lane) {
  int r = lane & 15, h = lane >> 4;
  const __bf16* p = base + r * ld + k0 + h * 8;
  v8bf lo = *(const v8bf*)(p);
  v8bf hi = *(const v8bf*)(p + 16);
  v16bf f;
#pragma unroll
  for (int i = 0; i < 8; i++) { f[i] = lo[i]; f[i + 8] = hi[i]; }
  return f;
}

// B operand: 32 (K) x 16 (N); source row n holds B column n (W[n,k]).
__device__ inline v16bf load_frag_b(const __bf16* base, int ld, int k0, int lane) {
  int r = lane & 15, h = lane >> 4;
  const __bf16* p = base + r * ld + k0 + h * 8;
  v8bf lo = *(const v8bf*)(p);
  v8bf hi = *(const v8bf*)(p + 16);
  v16bf f;
#pragma unroll
  for (int i = 0; i < 8; i++) { f[i] = lo[i]; f[i + 8] = hi[i]; }
  return f;
}

// ---- TDM: stage a [rows x 512] bf16 row-major tile into LDS ---------------
// D# per CDNA5 ISA ch.8: group0 = {count|flags, lds_addr, global_addr_lo,
// global_addr_hi|type=2}; group1 = {data_size, tensor dims/strides, tile dims}.

__device__ inline unsigned lds_off_of(const void* p) {
  return (unsigned)(unsigned long long)(uintptr_t)p;
}

__device__ inline void tdm_load_rows512_bf16(unsigned lds_off,
                                             const __bf16* gptr,
                                             unsigned rows) {
  unsigned long long ga = (unsigned long long)(uintptr_t)gptr;
  unsigned ga_lo = (unsigned)__builtin_amdgcn_readfirstlane((int)(unsigned)ga);
  unsigned ga_hi = (unsigned)__builtin_amdgcn_readfirstlane((int)(unsigned)(ga >> 32));
  unsigned lo    = (unsigned)__builtin_amdgcn_readfirstlane((int)lds_off);
  unsigned rws   = (unsigned)__builtin_amdgcn_readfirstlane((int)rows);

  v4u g0 = { 1u,                                  // count=1, user descriptor
             lo,                                  // lds_addr (bytes)
             ga_lo,                               // global_addr[31:0]
             (ga_hi & 0x01FFFFFFu) | (2u << 30) };// global_addr[56:32] | type=2
  v8i g1 = { (int)(1u << 16),      // wg_mask=0, data_size=1 (2 bytes)
             (int)(512u << 16),    // tensor_dim0 = 512 (bits 79:48, low half)
             (int)(4096u << 16),   // tensor_dim0 hi=0 | tensor_dim1 = 4096 lo
             (int)(512u << 16),    // tensor_dim1 hi=0 | tile_dim0 = 512
             (int)rws,             // tile_dim1 = rows, tile_dim2 = 0
             512,                  // tensor_dim0_stride = 512 (elements)
             0,                    // stride0 hi | tensor_dim1_stride lo16
             0x20 };               // tensor_dim1_stride[47:16] -> 2M elements
  v4i gz = { 0, 0, 0, 0 };
#if __clang_major__ >= 23
  v8i gz8 = { 0, 0, 0, 0, 0, 0, 0, 0 };
  __builtin_amdgcn_tensor_load_to_lds(g0, g1, gz, gz, gz8, 0);
#else
  __builtin_amdgcn_tensor_load_to_lds(g0, g1, gz, gz, 0);
#endif
}

// ---- Kernel 0: weight conversion f32 -> bf16 ------------------------------

__global__ __launch_bounds__(256) void convert_w_kernel(
    const float* __restrict__ a, const float* __restrict__ b,
    const float* __restrict__ c, const float* __restrict__ d,
    __bf16* __restrict__ oa, __bf16* __restrict__ ob,
    __bf16* __restrict__ oc, __bf16* __restrict__ od) {
  int i = blockIdx.x * 256 + threadIdx.x;
  if (i < CDIM * CDIM) {
    oa[i] = (__bf16)a[i];
    ob[i] = (__bf16)b[i];
    oc[i] = (__bf16)c[i];
    od[i] = (__bf16)d[i];
  }
}

// ---- Kernel 1: GroupNorm -> bf16 tokens [b, N, C] -------------------------

__global__ __launch_bounds__(256) void gn_kernel(
    const float* __restrict__ x, const float* __restrict__ gamma,
    const float* __restrict__ beta, __bf16* __restrict__ t) {
  int b = blockIdx.x >> 5;
  int g = blockIdx.x & 31;
  const float* xb = x + ((size_t)b * CDIM + g * 16) * NTOK;
  int tid = threadIdx.x;
  const int GSZ = 16 * NTOK;

  float s = 0.f, ss = 0.f;
  for (int i = tid; i < GSZ; i += 256) {
    float v = xb[i];
    s += v; ss += v * v;
  }
  __shared__ float rs[256], rq[256];
  rs[tid] = s; rq[tid] = ss;
  __syncthreads();
#pragma unroll
  for (int off = 128; off > 0; off >>= 1) {
    if (tid < off) { rs[tid] += rs[tid + off]; rq[tid] += rq[tid + off]; }
    __syncthreads();
  }
  float mean = rs[0] * (1.0f / GSZ);
  float var  = rq[0] * (1.0f / GSZ) - mean * mean;
  float inv  = rsqrtf(var + 1e-6f);

  for (int i = tid; i < GSZ; i += 256) {
    int cl = i >> 12;
    int n  = i & (NTOK - 1);
    int c  = g * 16 + cl;
    float v = (xb[i] - mean) * inv * gamma[c] + beta[c];
    t[((size_t)b * NTOK + n) * CDIM + c] = (__bf16)v;
  }
}

// ---- Kernel 2: fused QKV projection ---------------------------------------
// Block = 32 token rows (A-tile TDM-staged in LDS, shared by 8 waves).
// Each wave: 32x32 macro-tile x {Q,K,V} -> 12 f32 accumulators,
// 12 WMMAs per K-step off 2 A-frags (LDS) + 6 B-frags (global).

__global__ __launch_bounds__(256) void qkv_kernel(
    const __bf16* __restrict__ t,
    const __bf16* __restrict__ wq, const __bf16* __restrict__ wk,
    const __bf16* __restrict__ wv,
    const float* __restrict__ bq, const float* __restrict__ bk,
    const float* __restrict__ bv,
    __bf16* __restrict__ q, __bf16* __restrict__ k, __bf16* __restrict__ vt) {
  int tid = threadIdx.x, lane = tid & 31, w = tid >> 5;
  int bb  = blockIdx.x >> 8;          // batch
  int rem = blockIdx.x & 255;
  int nt2 = rem >> 1;                 // 128 row-groups of 32 tokens
  int cb  = rem & 1;                  // 2 column halves
  int r0  = nt2 * 32;
  int c0  = (cb * 8 + w) * 32;

  __shared__ __align__(16) __bf16 sA[32][CDIM];   // 32 KB
  const __bf16* abase = t + ((size_t)bb * NTOK + r0) * CDIM;
  if (tid < 32) {
    tdm_load_rows512_bf16(lds_off_of(&sA[0][0]), abase, 32);
    __builtin_amdgcn_s_wait_tensorcnt(0);
  }
  __syncthreads();

  const __bf16* wsrc[3] = { wq + (size_t)c0 * CDIM,
                            wk + (size_t)c0 * CDIM,
                            wv + (size_t)c0 * CDIM };
  v8f acc[3][2][2] = {};
#pragma unroll
  for (int k0 = 0; k0 < CDIM; k0 += 32) {
    v16bf a0 = load_frag_a(&sA[0][0],  CDIM, k0, lane);
    v16bf a1 = load_frag_a(&sA[16][0], CDIM, k0, lane);
#pragma unroll
    for (int m = 0; m < 3; m++) {
      v16bf b0 = load_frag_b(wsrc[m],             CDIM, k0, lane);
      v16bf b1 = load_frag_b(wsrc[m] + 16 * CDIM, CDIM, k0, lane);
      acc[m][0][0] = wmma_bf16(a0, b0, acc[m][0][0]);
      acc[m][0][1] = wmma_bf16(a0, b1, acc[m][0][1]);
      acc[m][1][0] = wmma_bf16(a1, b0, acc[m][1][0]);
      acc[m][1][1] = wmma_bf16(a1, b1, acc[m][1][1]);
    }
  }

  int cl = lane & 15, rh = 8 * (lane >> 4);
#pragma unroll
  for (int cs = 0; cs < 2; cs++) {
    int c = c0 + cs * 16 + cl;
    float biasq = bq[c], biask = bk[c], biasv = bv[c];
#pragma unroll
    for (int rs = 0; rs < 2; rs++) {
#pragma unroll
      for (int j = 0; j < 8; j++) {
        int n = r0 + rs * 16 + rh + j;
        q[((size_t)bb * NTOK + n) * CDIM + c]  = (__bf16)(acc[0][rs][cs][j] + biasq);
        k[((size_t)bb * NTOK + n) * CDIM + c]  = (__bf16)(acc[1][rs][cs][j] + biask);
        vt[((size_t)bb * CDIM + c) * NTOK + n] = (__bf16)(acc[2][rs][cs][j] + biasv);
      }
    }
  }
}

// ---- Kernel 3: flash attention --------------------------------------------
// Block = 16 query rows (Q-tile TDM-staged in LDS, reused by all 8 waves
// across all 32 key-tile iterations). Online softmax; P x V per 64-col slice.

__global__ __launch_bounds__(256) void attn_kernel(
    const __bf16* __restrict__ q, const __bf16* __restrict__ k,
    const __bf16* __restrict__ vt, __bf16* __restrict__ o) {
  int bb = blockIdx.x >> 8;
  int rt = blockIdx.x & 255;
  int tid = threadIdx.x;
  int lane = tid & 31;
  int w = tid >> 5;

  const __bf16* qbase = q + ((size_t)bb * NTOK + rt * 16) * CDIM;
  const __bf16* kbase = k + (size_t)bb * NTOK * CDIM;
  const __bf16* vbase = vt + (size_t)bb * CDIM * NTOK;

  __shared__ __align__(16) __bf16 sQ[16][CDIM];   // 16 KB, TDM-staged
  __shared__ __align__(16) float  sS[16][128];
  __shared__ __align__(16) __bf16 sP[16][128];
  __shared__ float red[16][16];
  __shared__ float sMax[16], sSum[16], sFac[16];

  if (tid < 32) {
    tdm_load_rows512_bf16(lds_off_of(&sQ[0][0]), qbase, 16);
    if (lane < 16) { sMax[lane] = -1e30f; sSum[lane] = 0.f; }
    __builtin_amdgcn_s_wait_tensorcnt(0);
  }
  v8f oacc[4] = {v8f{}, v8f{}, v8f{}, v8f{}};
  __syncthreads();

  for (int m0 = 0; m0 < NTOK; m0 += 128) {
    // prefetch next K/V tile into cache while we compute this one
    if (m0 + 128 < NTOK) {
      const __bf16* nk = kbase + (size_t)(m0 + 128) * CDIM;
      __builtin_prefetch(nk + (size_t)tid * 256, 0, 1);
      __builtin_prefetch(nk + (size_t)tid * 256 + 64, 0, 1);
      const __bf16* nv = vbase + (size_t)(tid * 2) * NTOK + (m0 + 128);
      __builtin_prefetch(nv, 0, 1);
    }

    // --- scores: each wave one 16x16 subtile over full K=512 (Q from LDS)
    v8f sacc = {};
    const __bf16* kb = kbase + (size_t)(m0 + 16 * w) * CDIM;
#pragma unroll
    for (int k0 = 0; k0 < CDIM; k0 += 32)
      sacc = wmma_bf16(load_frag_a(&sQ[0][0], CDIM, k0, lane),
                       load_frag_b(kb, CDIM, k0, lane), sacc);
    {
      int rb = 8 * (lane >> 4);
      int col = 16 * w + (lane & 15);
#pragma unroll
      for (int j = 0; j < 8; j++) sS[rb + j][col] = sacc[j] * ATTN_SCALE;
    }
    __syncthreads();

    // --- row max
    {
      int row = tid & 15, seg = tid >> 4;
      float mx = sS[row][seg * 8];
#pragma unroll
      for (int i = 1; i < 8; i++) mx = fmaxf(mx, sS[row][seg * 8 + i]);
      red[row][seg] = mx;
    }
    __syncthreads();
    if (tid < 16) {
      float mx = sMax[tid];
#pragma unroll
      for (int i = 0; i < 16; i++) mx = fmaxf(mx, red[tid][i]);
      float fac = __expf(sMax[tid] - mx);
      sFac[tid] = fac;
      sSum[tid] *= fac;
      sMax[tid] = mx;
    }
    __syncthreads();

    // --- P = exp(S - max) -> bf16 LDS; partial row sums
    {
      int row = tid & 15, seg = tid >> 4;
      float mx = sMax[row], ps = 0.f;
#pragma unroll
      for (int i = 0; i < 8; i++) {
        float p = __expf(sS[row][seg * 8 + i] - mx);
        sP[row][seg * 8 + i] = (__bf16)p;
        ps += p;
      }
      red[row][seg] = ps;
    }
    __syncthreads();
    if (tid < 16) {
      float s = 0.f;
#pragma unroll
      for (int i = 0; i < 16; i++) s += red[tid][i];
      sSum[tid] += s;
    }
    __syncthreads();

    // --- rescale O accumulators
    {
      int rb = 8 * (lane >> 4);
      float f0[8];
#pragma unroll
      for (int j = 0; j < 8; j++) f0[j] = sFac[rb + j];
#pragma unroll
      for (int ct = 0; ct < 4; ct++)
#pragma unroll
        for (int j = 0; j < 8; j++) oacc[ct][j] *= f0[j];
    }

    // --- O += P(16x128) x V(128 x 64-slice); wave w owns cols [64w, 64w+64)
#pragma unroll
    for (int k0 = 0; k0 < 128; k0 += 32) {
      v16bf a = load_frag_a(&sP[0][0], 128, k0, lane);
#pragma unroll
      for (int ct = 0; ct < 4; ct++) {
        const __bf16* vb = vbase + (size_t)(64 * w + 16 * ct) * NTOK + m0;
        oacc[ct] = wmma_bf16(a, load_frag_b(vb, NTOK, k0, lane), oacc[ct]);
      }
    }
    __syncthreads();
  }

  // --- finalize: O /= rowsum, store bf16 [b,N,C]
  {
    int rb = 8 * (lane >> 4);
    float inv[8];
#pragma unroll
    for (int j = 0; j < 8; j++) inv[j] = 1.0f / sSum[rb + j];
    __bf16* ob = o + ((size_t)bb * NTOK + rt * 16) * CDIM;
#pragma unroll
    for (int ct = 0; ct < 4; ct++) {
      int col = 64 * w + 16 * ct + (lane & 15);
#pragma unroll
      for (int j = 0; j < 8; j++)
        ob[(size_t)(rb + j) * CDIM + col] = (__bf16)(oacc[ct][j] * inv[j]);
    }
  }
}

// ---- Kernel 4: output projection + bias + residual ------------------------
// Same macro-tiling as QKV (32x32 per wave, A-tile TDM-staged in LDS).

__global__ __launch_bounds__(256) void proj_kernel(
    const __bf16* __restrict__ o, const __bf16* __restrict__ wo,
    const float* __restrict__ bo, const float* __restrict__ x,
    float* __restrict__ out) {
  int tid = threadIdx.x, lane = tid & 31, w = tid >> 5;
  int bb  = blockIdx.x >> 8;
  int rem = blockIdx.x & 255;
  int nt2 = rem >> 1;
  int cb  = rem & 1;
  int r0  = nt2 * 32;
  int c0  = (cb * 8 + w) * 32;

  __shared__ __align__(16) __bf16 sA[32][CDIM];
  const __bf16* abase = o + ((size_t)bb * NTOK + r0) * CDIM;
  if (tid < 32) {
    tdm_load_rows512_bf16(lds_off_of(&sA[0][0]), abase, 32);
    __builtin_amdgcn_s_wait_tensorcnt(0);
  }
  __syncthreads();

  const __bf16* wb = wo + (size_t)c0 * CDIM;
  v8f acc[2][2] = {};
#pragma unroll
  for (int k0 = 0; k0 < CDIM; k0 += 32) {
    v16bf a0 = load_frag_a(&sA[0][0],  CDIM, k0, lane);
    v16bf a1 = load_frag_a(&sA[16][0], CDIM, k0, lane);
    v16bf b0 = load_frag_b(wb,             CDIM, k0, lane);
    v16bf b1 = load_frag_b(wb + 16 * CDIM, CDIM, k0, lane);
    acc[0][0] = wmma_bf16(a0, b0, acc[0][0]);
    acc[0][1] = wmma_bf16(a0, b1, acc[0][1]);
    acc[1][0] = wmma_bf16(a1, b0, acc[1][0]);
    acc[1][1] = wmma_bf16(a1, b1, acc[1][1]);
  }

  int cl = lane & 15, rh = 8 * (lane >> 4);
#pragma unroll
  for (int cs = 0; cs < 2; cs++) {
    int c = c0 + cs * 16 + cl;
    float bias = bo[c];
    size_t base = ((size_t)bb * CDIM + c) * NTOK;
#pragma unroll
    for (int rs = 0; rs < 2; rs++) {
#pragma unroll
      for (int j = 0; j < 8; j++) {
        size_t idx = base + r0 + rs * 16 + rh + j;
        out[idx] = acc[rs][cs][j] + bias + x[idx];
      }
    }
  }
}

// ---- launch ---------------------------------------------------------------

extern "C" void kernel_launch(void* const* d_in, const int* in_sizes, int n_in,
                              void* d_out, int out_size, void* d_ws, size_t ws_size,
                              hipStream_t stream) {
  (void)in_sizes; (void)n_in; (void)out_size; (void)ws_size;
  const float* x     = (const float*)d_in[0];
  const float* gamma = (const float*)d_in[1];
  const float* beta  = (const float*)d_in[2];
  const float* wq_w  = (const float*)d_in[3];
  const float* wq_b  = (const float*)d_in[4];
  const float* wk_w  = (const float*)d_in[5];
  const float* wk_b  = (const float*)d_in[6];
  const float* wv_w  = (const float*)d_in[7];
  const float* wv_b  = (const float*)d_in[8];
  const float* out_w = (const float*)d_in[9];
  const float* out_b = (const float*)d_in[10];
  float* out = (float*)d_out;

  char* ws = (char*)d_ws;
  const size_t TSZ = (size_t)NB * NTOK * CDIM * sizeof(__bf16); // 16 MB
  __bf16* t_bf  = (__bf16*)(ws);
  __bf16* q_bf  = (__bf16*)(ws + TSZ);
  __bf16* k_bf  = (__bf16*)(ws + 2 * TSZ);
  __bf16* vt_bf = (__bf16*)(ws + 3 * TSZ);
  __bf16* o_bf  = (__bf16*)(ws + 4 * TSZ);
  __bf16* wq_bf = (__bf16*)(ws + 5 * TSZ);
  __bf16* wk_bf = wq_bf + (size_t)CDIM * CDIM;
  __bf16* wv_bf = wq_bf + (size_t)2 * CDIM * CDIM;
  __bf16* wo_bf = wq_bf + (size_t)3 * CDIM * CDIM;

  convert_w_kernel<<<(CDIM * CDIM + 255) / 256, 256, 0, stream>>>(
      wq_w, wk_w, wv_w, out_w, wq_bf, wk_bf, wv_bf, wo_bf);
  gn_kernel<<<NB * 32, 256, 0, stream>>>(x, gamma, beta, t_bf);
  qkv_kernel<<<NB * 256, 256, 0, stream>>>(
      t_bf, wq_bf, wk_bf, wv_bf, wq_b, wk_b, wv_b, q_bf, k_bf, vt_bf);
  attn_kernel<<<NB * 256, 256, 0, stream>>>(q_bf, k_bf, vt_bf, o_bf);
  proj_kernel<<<NB * 256, 256, 0, stream>>>(
      o_bf, wo_bf, out_b, x, out);
}
